// RecurrentHLDGNNetwork_78030965834387
// MI455X (gfx1250) — compile-verified, matrix operand order checked by reading, not run on hardware
//
#include <hip/hip_runtime.h>

typedef __attribute__((ext_vector_type(16))) _Float16 v16h;
typedef __attribute__((ext_vector_type(8)))  _Float16 v8h;
typedef __attribute__((ext_vector_type(8)))  float    v8f;

#define N_NODES   65536
#define N_GRAPHS  256
#define NPG       256
#define N_EDGES   1048576
#define E_TOT     (N_EDGES + N_NODES)
#define IN_DIM    64
#define HEADS     4
#define HID       64
#define LATENT    256
#define BSZ       16
#define HORIZON   16
#define ACTIONS   10
#define MLP_HID   128

// ---------------- WMMA fragment swizzle (ISA 7.12.2, wave32) ----------------
// A 16x32 (and B^T) fragment: lane = m + 16*L, half index h = 8*c + r,
// where klocal = 16*c + 8*L + r.  One fragment = 32 lanes * 16 halves = 1KB.
__device__ __forceinline__ int frag_off(int m, int kl) {
  int L = (kl >> 3) & 1, c = (kl >> 4) & 1, r = kl & 7;
  return ((m + (L << 4)) << 4) + (c << 3) + r;
}

// Fragment load: 32 contiguous bytes per lane -> 2x ds/global_load_b128.
__device__ __forceinline__ v16h load_frag(const _Float16* fragbase, int lane) {
  return *(const v16h*)(fragbase + lane * 16);
}

// D/gate storage: per-lane 8 halves contiguous (one b128 store).
// element (row b, col j): lane = (j&15) + 16*(b>>3), slot = b&7.
__device__ __forceinline__ int g_off(int b, int j) {
  return ((((j >> 4) << 5) + (j & 15) + ((b >> 3) << 4)) << 3) + (b & 7);
}

__device__ __forceinline__ float sigmoidf(float x) {
  return 1.0f / (1.0f + __expf(-x));
}

__device__ __forceinline__ void atomicMaxFloat(float* addr, float v) {
  if (v >= 0.0f) atomicMax((int*)addr, __float_as_int(v));
  else           atomicMin((unsigned int*)addr, __float_as_uint(v));
}

// -------------------- K1: xl = x@Wl, xr = x@Wr via WMMA --------------------
// block = 256 threads (8 waves) -> 128 rows; grid = 512.
__global__ void k_transform(const float* __restrict__ x,
                            const float* __restrict__ Wl,
                            const float* __restrict__ Wr,
                            float* __restrict__ xl, float* __restrict__ xr) {
  __shared__ __align__(32) _Float16 sXf[8 * 2 * 512];   // (rowtile,kstep) frags, 16KB
  __shared__ __align__(32) _Float16 sWf[16 * 2 * 512];  // (ntile,kstep) frags, 32KB
  int tid = threadIdx.x, lane = tid & 31, wave = tid >> 5;
  int rowBase = blockIdx.x * 128;

  // stage x swizzled: coalesced float4 reads, contiguous 4-half frag stores
  const float4* x4 = (const float4*)&x[(size_t)rowBase * IN_DIM];
#pragma unroll
  for (int i4 = tid; i4 < 128 * IN_DIM / 4; i4 += 256) {
    float4 v = x4[i4];
    int row = i4 >> 4, k0 = (i4 & 15) * 4;
    _Float16* d = &sXf[(((row >> 4) << 1) + (k0 >> 5)) * 512 + frag_off(row & 15, k0 & 31)];
    d[0] = (_Float16)v.x; d[1] = (_Float16)v.y;
    d[2] = (_Float16)v.z; d[3] = (_Float16)v.w;
  }
  __syncthreads();

  v16h a0 = load_frag(&sXf[(wave * 2 + 0) * 512], lane);
  v16h a1 = load_frag(&sXf[(wave * 2 + 1) * 512], lane);

  for (int side = 0; side < 2; ++side) {
    const float* W = side ? Wr : Wl;
    float*       O = side ? xr : xl;
    __syncthreads();
    const float4* W4 = (const float4*)W;
    for (int i4 = tid; i4 < IN_DIM * LATENT / 4; i4 += 256) {
      float4 v = W4[i4];
      int k = i4 >> 6, n0 = (i4 & 63) * 4;
      _Float16* d = &sWf[(((n0 >> 4) << 1) + (k >> 5)) * 512 + frag_off(n0 & 15, k & 31)];
      d[0]  = (_Float16)v.x; d[16] = (_Float16)v.y;
      d[32] = (_Float16)v.z; d[48] = (_Float16)v.w;
    }
    __syncthreads();
#pragma unroll 4
    for (int nt = 0; nt < 16; ++nt) {
      v8f acc = {};
      v16h b0 = load_frag(&sWf[(nt * 2 + 0) * 512], lane);
      v16h b1 = load_frag(&sWf[(nt * 2 + 1) * 512], lane);
      acc = __builtin_amdgcn_wmma_f32_16x16x32_f16(false, a0, false, b0,
                                                   (short)0, acc, false, false);
      acc = __builtin_amdgcn_wmma_f32_16x16x32_f16(false, a1, false, b1,
                                                   (short)0, acc, false, false);
      int col = nt * 16 + (lane & 15);
      int rr  = (lane >> 4) << 3;
#pragma unroll
      for (int r = 0; r < 8; ++r)
        O[(size_t)(rowBase + wave * 16 + rr + r) * LATENT + col] = acc[r];
    }
  }
}

// -------------------- K2: init segment-max / segment-sum -------------------
__global__ void k_init_stats(float* __restrict__ m, float* __restrict__ s) {
  int i = blockIdx.x * blockDim.x + threadIdx.x;
  if (i < N_NODES * HEADS) { m[i] = -3.0e38f; s[i] = 0.0f; }
}

// -------------------- K3: wave-per-edge attention logits + segment max -----
__global__ void k_logits(const float* __restrict__ xl,
                         const float* __restrict__ xr,
                         const int* __restrict__ eidx,
                         const float* __restrict__ att,
                         float* __restrict__ lp, float* __restrict__ mM) {
  __shared__ float sAtt[LATENT];
  int tid = threadIdx.x, lane = tid & 31;
  if (tid < LATENT) sAtt[tid] = att[tid];
  __syncthreads();

  int e = blockIdx.x * 8 + (tid >> 5);
  if (e >= E_TOT) return;
  int src, dst;
  if (e < N_EDGES) { src = eidx[e]; dst = eidx[N_EDGES + e]; }
  else             { src = dst = e - N_EDGES; }

  int d0 = lane * 8;                        // 8 dims per lane, head = lane>>3
  const float4* a4 = (const float4*)&xl[(size_t)src * LATENT + d0];
  const float4* b4 = (const float4*)&xr[(size_t)dst * LATENT + d0];
  float4 a0 = a4[0], a1 = a4[1], b0 = b4[0], b1 = b4[1];
  float p = 0.0f, t;
  t = a0.x + b0.x; t = t > 0.f ? t : 0.2f * t; p += t * sAtt[d0 + 0];
  t = a0.y + b0.y; t = t > 0.f ? t : 0.2f * t; p += t * sAtt[d0 + 1];
  t = a0.z + b0.z; t = t > 0.f ? t : 0.2f * t; p += t * sAtt[d0 + 2];
  t = a0.w + b0.w; t = t > 0.f ? t : 0.2f * t; p += t * sAtt[d0 + 3];
  t = a1.x + b1.x; t = t > 0.f ? t : 0.2f * t; p += t * sAtt[d0 + 4];
  t = a1.y + b1.y; t = t > 0.f ? t : 0.2f * t; p += t * sAtt[d0 + 5];
  t = a1.z + b1.z; t = t > 0.f ? t : 0.2f * t; p += t * sAtt[d0 + 6];
  t = a1.w + b1.w; t = t > 0.f ? t : 0.2f * t; p += t * sAtt[d0 + 7];
#pragma unroll
  for (int off = 1; off < 8; off <<= 1) p += __shfl_xor(p, off, 32);
  if ((lane & 7) == 0) {
    int h = lane >> 3;
    lp[(size_t)e * HEADS + h] = p;
    atomicMaxFloat(&mM[dst * HEADS + h], p);
  }
}

// -------------------- K4: p = exp(logit - max), segment sum ----------------
__global__ void k_expsum(const int* __restrict__ eidx,
                         float* __restrict__ lp,
                         const float* __restrict__ mM,
                         float* __restrict__ sS) {
  int e = blockIdx.x * 256 + threadIdx.x;
  if (e >= E_TOT) return;
  int dst = (e < N_EDGES) ? eidx[N_EDGES + e] : (e - N_EDGES);
#pragma unroll
  for (int h = 0; h < HEADS; ++h) {
    float p = __expf(lp[(size_t)e * HEADS + h] - mM[dst * HEADS + h]);
    lp[(size_t)e * HEADS + h] = p;
    atomicAdd(&sS[dst * HEADS + h], p);
  }
}

// -------------------- K4b: zero the aggregation buffer ---------------------
__global__ void k_zero(float* __restrict__ a) {
  size_t i = (size_t)blockIdx.x * 256 + threadIdx.x;
  if (i < (size_t)N_NODES * LATENT) a[i] = 0.0f;
}

// -------------------- K5: out[dst] += alpha * xl[src] (wave per edge) ------
__global__ void k_aggregate(const int* __restrict__ eidx,
                            const float* __restrict__ lp,
                            const float* __restrict__ sS,
                            const float* __restrict__ xl,
                            float* __restrict__ agg) {
  int tid = threadIdx.x, lane = tid & 31;
  int e = blockIdx.x * 8 + (tid >> 5);
  if (e >= E_TOT) return;
  int src, dst;
  if (e < N_EDGES) { src = eidx[e]; dst = eidx[N_EDGES + e]; }
  else             { src = dst = e - N_EDGES; }
  int h = lane >> 3;
  float alpha = lp[(size_t)e * HEADS + h] / sS[dst * HEADS + h];
  int d0 = lane * 8;
  const float4* xs = (const float4*)&xl[(size_t)src * LATENT + d0];
  float4 v0 = xs[0], v1 = xs[1];
  float* o = &agg[(size_t)dst * LATENT + d0];
  atomicAdd(&o[0], alpha * v0.x); atomicAdd(&o[1], alpha * v0.y);
  atomicAdd(&o[2], alpha * v0.z); atomicAdd(&o[3], alpha * v0.w);
  atomicAdd(&o[4], alpha * v1.x); atomicAdd(&o[5], alpha * v1.y);
  atomicAdd(&o[6], alpha * v1.z); atomicAdd(&o[7], alpha * v1.w);
}

// -------------------- K6: bias + relu + per-graph max pool -----------------
__global__ void k_pool(const float* __restrict__ agg,
                       const float* __restrict__ bias,
                       float* __restrict__ pooled) {
  int g = blockIdx.x, c = threadIdx.x;
  float b = bias[c];
  float mx = 0.0f;
  const float* base = agg + (size_t)g * NPG * LATENT + c;
#pragma unroll 4
  for (int i = 0; i < NPG; ++i) {
    float v = base[(size_t)i * LATENT] + b;
    v = v > 0.f ? v : 0.f;
    mx = fmaxf(mx, v);
  }
  pooled[g * LATENT + c] = mx;
}

// ---- K6b: pack [W_ih; W_hh] f16, fragment-major [64 ntiles][16 kk][1KB] ---
// output-contiguous: each thread emits 2 consecutive halves (one b32 store).
__global__ void k_prep_w(const float* __restrict__ W_ih,
                         const float* __restrict__ W_hh,
                         _Float16* __restrict__ Wc) {
  int o2 = blockIdx.x * 256 + threadIdx.x;     // over 512*1024/2
  if (o2 >= 512 * 1024 / 2) return;
  int o = o2 * 2;
  int fragidx = o >> 9, within = o & 511;
  int lane = within >> 4, hh = within & 15;    // hh even
  int nt = fragidx >> 4, kk = fragidx & 15;
  int j = nt * 16 + (lane & 15);
  int k = kk * 32 + ((hh >> 3) << 4) + ((lane >> 4) << 3) + (hh & 7);
  const float* srcp = (k < LATENT) ? &W_ih[j * LATENT + k]
                                   : &W_hh[j * LATENT + (k - LATENT)];
  Wc[o]     = (_Float16)srcp[0];
  Wc[o + 1] = (_Float16)srcp[1];
}

// -------------------- K7: LSTM (WMMA gates) + dueling heads ----------------
// single block, 256 threads (8 waves). c in regs, h inside f16 A fragments.
__global__ void k_lstm_head(const float* __restrict__ pooled,
                            const _Float16* __restrict__ Wc,
                            const float* __restrict__ b_ih,
                            const float* __restrict__ b_hh,
                            const float* __restrict__ Wq1, const float* __restrict__ bq1,
                            const float* __restrict__ Wq2, const float* __restrict__ bq2,
                            const float* __restrict__ Wv1, const float* __restrict__ bv1,
                            const float* __restrict__ Wv2, const float* __restrict__ bv2,
                            float* __restrict__ out) {
  __shared__ __align__(32) _Float16 sAf[16 * 512];    // A frags [kk][1KB], 16KB
  __shared__ __align__(32) _Float16 sGf[64 * 256];    // gate frags (v8h/lane), 32KB
  int tid = threadIdx.x, lane = tid & 31, wave = tid >> 5;
  int u = tid;                                        // hidden index 0..255

  float creg[BSZ];
#pragma unroll
  for (int b = 0; b < BSZ; ++b) creg[b] = 0.0f;
  float bI = b_ih[u]       + b_hh[u];
  float bF = b_ih[256 + u] + b_hh[256 + u];
  float bG = b_ih[512 + u] + b_hh[512 + u];
  float bO = b_ih[768 + u] + b_hh[768 + u];

  for (int i = tid; i < 16 * 512; i += 256) sAf[i] = (_Float16)0.0f;  // h0 = 0

  for (int t = 0; t < HORIZON; ++t) {
    // x_t into A fragments (kk 0..7): float4 reads, 4 contiguous halves store
    for (int i4 = tid; i4 < BSZ * LATENT / 4; i4 += 256) {
      int b = i4 >> 6, k0 = (i4 & 63) * 4;
      float4 v = *(const float4*)&pooled[(b * HORIZON + t) * LATENT + k0];
      _Float16* d = &sAf[(k0 >> 5) * 512 + frag_off(b, k0 & 31)];
      d[0] = (_Float16)v.x; d[1] = (_Float16)v.y;
      d[2] = (_Float16)v.z; d[3] = (_Float16)v.w;
    }
    __syncthreads();
    // gates = A[16x512] @ Wc[512x1024]; 8 waves x 8 n-tiles, B128 frag loads
    for (int i = 0; i < 8; ++i) {
      int nt = wave * 8 + i;
      v8f acc = {};
#pragma unroll 4
      for (int kk = 0; kk < 16; ++kk) {
        v16h a = load_frag(&sAf[kk * 512], lane);
        v16h b = load_frag(&Wc[(nt * 16 + kk) * 512], lane);
        acc = __builtin_amdgcn_wmma_f32_16x16x32_f16(false, a, false, b,
                                                     (short)0, acc, false, false);
      }
      v8h hacc;
#pragma unroll
      for (int r = 0; r < 8; ++r) hacc[r] = (_Float16)acc[r];
      *(v8h*)&sGf[(nt * 32 + lane) * 8] = hacc;       // one ds_store_b128
    }
    __syncthreads();
    // elementwise LSTM cell; h written back into A fragments (kk 8..15)
#pragma unroll 4
    for (int b = 0; b < BSZ; ++b) {
      float gi = (float)sGf[g_off(b, u)]       + bI;
      float gf = (float)sGf[g_off(b, 256 + u)] + bF;
      float gg = (float)sGf[g_off(b, 512 + u)] + bG;
      float go = (float)sGf[g_off(b, 768 + u)] + bO;
      float c  = sigmoidf(gf) * creg[b] + sigmoidf(gi) * tanhf(gg);
      creg[b]  = c;
      int kg = 256 + u;
      sAf[(kg >> 5) * 512 + frag_off(b, kg & 31)] = (_Float16)(sigmoidf(go) * tanhf(c));
    }
  }
  __syncthreads();

  // ---- dueling heads on h_last (f16 in A fragments, kk 8..15) ----
  float* q1 = (float*)sGf;              // 16x128
  float* v1 = q1 + BSZ * MLP_HID;       // 16x128
  for (int i = tid; i < BSZ * MLP_HID; i += 256) {
    int b = i / MLP_HID, j = i % MLP_HID;
    float aq = bq1[j], av = bv1[j];
    for (int k = 0; k < LATENT; ++k) {
      int kg = 256 + k;
      float h = (float)sAf[(kg >> 5) * 512 + frag_off(b, kg & 31)];
      aq += h * Wq1[k * MLP_HID + j];
      av += h * Wv1[k * MLP_HID + j];
    }
    q1[i] = fmaxf(aq, 0.0f);
    v1[i] = fmaxf(av, 0.0f);
  }
  __syncthreads();
  float* qb = v1 + BSZ * MLP_HID;       // 16x10
  float* vb = qb + BSZ * ACTIONS;       // 16
  if (tid < BSZ * ACTIONS) {
    int b = tid / ACTIONS, a = tid % ACTIONS;
    float acc = bq2[a];
    for (int j = 0; j < MLP_HID; ++j) acc += q1[b * MLP_HID + j] * Wq2[j * ACTIONS + a];
    qb[tid] = acc;
  }
  if (tid < BSZ) {
    float acc = bv2[0];
    for (int j = 0; j < MLP_HID; ++j) acc += v1[tid * MLP_HID + j] * Wv2[j];
    vb[tid] = acc;
  }
  __syncthreads();
  if (tid < BSZ * ACTIONS) {
    int b = tid / ACTIONS;
    float mean = 0.0f;
    for (int a = 0; a < ACTIONS; ++a) mean += qb[b * ACTIONS + a];
    mean *= (1.0f / ACTIONS);
    out[tid] = qb[tid] - mean + vb[b];
  }
}

// ---------------------------------------------------------------------------
extern "C" void kernel_launch(void* const* d_in, const int* in_sizes, int n_in,
                              void* d_out, int out_size, void* d_ws, size_t ws_size,
                              hipStream_t stream) {
  const float* x        = (const float*)d_in[0];
  const int*   eidx     = (const int*)d_in[1];
  const float* Wl       = (const float*)d_in[3];
  const float* Wr       = (const float*)d_in[4];
  const float* att      = (const float*)d_in[5];
  const float* bias_gat = (const float*)d_in[6];
  const float* W_ih     = (const float*)d_in[7];
  const float* W_hh     = (const float*)d_in[8];
  const float* b_ih     = (const float*)d_in[9];
  const float* b_hh     = (const float*)d_in[10];
  const float* Wq1      = (const float*)d_in[11];
  const float* bq1      = (const float*)d_in[12];
  const float* Wq2      = (const float*)d_in[13];
  const float* bq2      = (const float*)d_in[14];
  const float* Wv1      = (const float*)d_in[15];
  const float* bv1      = (const float*)d_in[16];
  const float* Wv2      = (const float*)d_in[17];
  const float* bv2      = (const float*)d_in[18];

  char* p = (char*)d_ws;
  float* xl = (float*)p;     p += (size_t)N_NODES * LATENT * sizeof(float);
  float* xr = (float*)p;     p += (size_t)N_NODES * LATENT * sizeof(float); // reused as agg
  float* lp = (float*)p;     p += (size_t)E_TOT * HEADS * sizeof(float);
  float* mM = (float*)p;     p += (size_t)N_NODES * HEADS * sizeof(float);
  float* sS = (float*)p;     p += (size_t)N_NODES * HEADS * sizeof(float);
  float* pooled = (float*)p; p += (size_t)N_GRAPHS * LATENT * sizeof(float);
  _Float16* Wcomb = (_Float16*)p;

  k_transform<<<N_NODES / 128, 256, 0, stream>>>(x, Wl, Wr, xl, xr);
  k_init_stats<<<(N_NODES * HEADS + 255) / 256, 256, 0, stream>>>(mM, sS);
  k_prep_w<<<(512 * 1024 / 2) / 256, 256, 0, stream>>>(W_ih, W_hh, Wcomb);
  k_logits<<<(E_TOT + 7) / 8, 256, 0, stream>>>(xl, xr, eidx, att, lp, mM);
  k_expsum<<<(E_TOT + 255) / 256, 256, 0, stream>>>(eidx, lp, mM, sS);
  k_zero<<<(N_NODES * LATENT) / 256, 256, 0, stream>>>(xr);       // xr -> agg
  k_aggregate<<<(E_TOT + 7) / 8, 256, 0, stream>>>(eidx, lp, sS, xl, xr);
  k_pool<<<N_GRAPHS, 256, 0, stream>>>(xr, bias_gat, pooled);
  k_lstm_head<<<1, 256, 0, stream>>>(pooled, Wcomb, b_ih, b_hh,
                                     Wq1, bq1, Wq2, bq2, Wv1, bv1, Wv2, bv2,
                                     (float*)d_out);
}
